// ProposalModule_53558242181148
// MI455X (gfx1250) — compile-verified, matrix operand order checked by reading, not run on hardware
//
#include <hip/hip_runtime.h>
#include <hip/hip_bf16.h>

typedef _Float16 half_t;
typedef __attribute__((ext_vector_type(8)))  _Float16 v8h;
typedef __attribute__((ext_vector_type(16))) _Float16 v16h;
typedef __attribute__((ext_vector_type(8)))  float    v8f;

union V16 { v16h v; v8h h[2]; };

#define EPS_BN 1e-5f
#define RAD2   25.0f

// ---------------------------------------------------------------------------
// Shared WMMA building blocks.
// A fragment (16x32 f16): lane L, element h -> M = L%16, K = (h/8)*16 + (L/16)*8 + h%8
// B fragment (32x16 f16): lane L, element h -> N = L%16, K = same formula
// C/D (16x16 f32): lane L, element r -> M = r + (L/16)*8, N = L%16
// ---------------------------------------------------------------------------
__device__ __forceinline__ void gemm_tiles(const half_t* __restrict__ Wg, int wstride,
                                           int kchunks, const half_t* Bl, int bs,
                                           int wbase, int lane, v8f* acc)
{
    const int group = lane >> 4;
    const int l16   = lane & 15;
    for (int kc = 0; kc < kchunks; ++kc) {
        const half_t* ap = Wg + (wbase + l16) * wstride + kc * 32 + group * 8;
        V16 A;
        A.h[0] = *(const v8h*)ap;
        A.h[1] = *(const v8h*)(ap + 16);
#pragma unroll
        for (int nt = 0; nt < 8; ++nt) {
            const half_t* bp = Bl + (nt * 16 + l16) * bs + kc * 32 + group * 8;
            V16 Bf;
            Bf.h[0] = *(const v8h*)bp;
            Bf.h[1] = *(const v8h*)(bp + 16);
            acc[nt] = __builtin_amdgcn_wmma_f32_16x16x32_f16(
                false, A.v, false, Bf.v, (short)0, acc[nt], false, false);
        }
    }
}

__device__ __forceinline__ void epi_store(const v8f* acc, const float* al, const float* be,
                                          half_t* Ol, int os, int wbase, int lane)
{
    const int group = lane >> 4;
    const int l16   = lane & 15;
    const int ch0   = wbase + group * 8;
#pragma unroll
    for (int nt = 0; nt < 8; ++nt) {
        v8h o;
#pragma unroll
        for (int r = 0; r < 8; ++r) {
            float y = acc[nt][r] * al[ch0 + r] + be[ch0 + r];
            y = fmaxf(y, 0.0f);
            o[r] = (half_t)y;
        }
        *(v8h*)(Ol + (nt * 16 + l16) * os + ch0) = o;
    }
}

__device__ __forceinline__ v8f vzero8()
{
    v8f z = {0.f, 0.f, 0.f, 0.f, 0.f, 0.f, 0.f, 0.f};
    return z;
}

// ---------------------------------------------------------------------------
// K0a: transpose+convert features (B,C=256,K=1024) f32 -> featT (B,K,C) f16
// ---------------------------------------------------------------------------
__global__ __launch_bounds__(256) void k0_featT(const float* __restrict__ f,
                                                half_t* __restrict__ ft)
{
    int i = blockIdx.x * 256 + threadIdx.x;   // 0 .. 4*1024*256-1
    int b = i >> 18;
    int r = i & 262143;
    int k = r >> 8;
    int c = r & 255;
    ft[i] = (half_t)f[(b << 18) + (c << 10) + k];
}

// ---------------------------------------------------------------------------
// K0b: convert/pad weights, fold BN into alpha/beta.
// w0p column order: c' in [0,256) = input row c'+3 (features),
//                   c' in [256,259) = xyz rows 0..2, c' in [259,288) = 0.
// ---------------------------------------------------------------------------
__global__ __launch_bounds__(256) void k0_misc(
    const float* __restrict__ w0, const float* __restrict__ w1, const float* __restrict__ w2,
    const float* __restrict__ pw0, const float* __restrict__ pw1, const float* __restrict__ pw2,
    const float* __restrict__ pb2,
    const float* __restrict__ g0, const float* __restrict__ b0, const float* __restrict__ m0, const float* __restrict__ v0,
    const float* __restrict__ g1, const float* __restrict__ b1, const float* __restrict__ m1, const float* __restrict__ v1,
    const float* __restrict__ g2, const float* __restrict__ b2, const float* __restrict__ m2, const float* __restrict__ v2,
    const float* __restrict__ pg0, const float* __restrict__ pb0, const float* __restrict__ pm0, const float* __restrict__ pv0,
    const float* __restrict__ pg1, const float* __restrict__ pb1, const float* __restrict__ pm1, const float* __restrict__ pv1,
    const float* __restrict__ ms,
    half_t* __restrict__ w0p, half_t* __restrict__ w1h, half_t* __restrict__ w2h,
    half_t* __restrict__ pw0h, half_t* __restrict__ pw1h, half_t* __restrict__ pw2p,
    float* __restrict__ ab, float* __restrict__ pb2p, float* __restrict__ msd)
{
    const int t = threadIdx.x;
    for (int i = t; i < 128 * 288; i += 256) {
        int o = i / 288, c = i % 288;
        float val = (c < 256) ? w0[o * 259 + c + 3]
                              : ((c < 259) ? w0[o * 259 + (c - 256)] : 0.0f);
        w0p[i] = (half_t)val;
    }
    for (int i = t; i < 16384; i += 256) {
        w1h[i]  = (half_t)w1[i];
        w2h[i]  = (half_t)w2[i];
        pw0h[i] = (half_t)pw0[i];
        pw1h[i] = (half_t)pw1[i];
    }
    for (int i = t; i < 96 * 128; i += 256) {
        int o = i / 128, c = i % 128;
        pw2p[i] = (half_t)((o < 93) ? pw2[o * 128 + c] : 0.0f);
    }
    if (t < 128) {
        float a;
        a = g0[t]  * rsqrtf(v0[t]  + EPS_BN); ab[0 * 256 + t] = a; ab[0 * 256 + 128 + t] = b0[t]  - m0[t]  * a;
        a = g1[t]  * rsqrtf(v1[t]  + EPS_BN); ab[1 * 256 + t] = a; ab[1 * 256 + 128 + t] = b1[t]  - m1[t]  * a;
        a = g2[t]  * rsqrtf(v2[t]  + EPS_BN); ab[2 * 256 + t] = a; ab[2 * 256 + 128 + t] = b2[t]  - m2[t]  * a;
        a = pg0[t] * rsqrtf(pv0[t] + EPS_BN); ab[3 * 256 + t] = a; ab[3 * 256 + 128 + t] = pb0[t] - pm0[t] * a;
        a = pg1[t] * rsqrtf(pv1[t] + EPS_BN); ab[4 * 256 + t] = a; ab[4 * 256 + 128 + t] = pb1[t] - pm1[t] * a;
    }
    if (t < 96) pb2p[t] = (t < 93) ? pb2[t] : 0.0f;
    if (t < 54) msd[t] = ms[t];
}

// ---------------------------------------------------------------------------
// K1: farthest point sampling; one wave per batch, all state in registers.
// ---------------------------------------------------------------------------
__global__ __launch_bounds__(32) void k1_fps(const float* __restrict__ xyz,
                                             int* __restrict__ fpsi)
{
    const int b    = blockIdx.x;
    const int lane = threadIdx.x;
    const float* xb = xyz + b * 3072;
    float xs[32], ys[32], zs[32], dd[32];
#pragma unroll
    for (int j = 0; j < 32; ++j) {
        int k = lane * 32 + j;
        xs[j] = xb[k * 3 + 0];
        ys[j] = xb[k * 3 + 1];
        zs[j] = xb[k * 3 + 2];
        dd[j] = 1e10f;
    }
    int last = 0;
    if (lane == 0) fpsi[b * 256] = 0;
    for (int it = 1; it < 256; ++it) {
        float lx = xb[last * 3 + 0], ly = xb[last * 3 + 1], lz = xb[last * 3 + 2];
        float bv = -1.0f;
        int   bi = 0;
#pragma unroll
        for (int j = 0; j < 32; ++j) {
            float dx = xs[j] - lx, dy = ys[j] - ly, dz = zs[j] - lz;
            float d  = dx * dx + dy * dy + dz * dz;
            dd[j] = fminf(dd[j], d);
            if (dd[j] > bv) { bv = dd[j]; bi = lane * 32 + j; }
        }
        float ov; int oi;
        ov = __shfl_xor(bv, 16, 32); oi = __shfl_xor(bi, 16, 32);
        if (ov > bv || (ov == bv && oi < bi)) { bv = ov; bi = oi; }
        ov = __shfl_xor(bv, 8, 32);  oi = __shfl_xor(bi, 8, 32);
        if (ov > bv || (ov == bv && oi < bi)) { bv = ov; bi = oi; }
        ov = __shfl_xor(bv, 4, 32);  oi = __shfl_xor(bi, 4, 32);
        if (ov > bv || (ov == bv && oi < bi)) { bv = ov; bi = oi; }
        ov = __shfl_xor(bv, 2, 32);  oi = __shfl_xor(bi, 2, 32);
        if (ov > bv || (ov == bv && oi < bi)) { bv = ov; bi = oi; }
        ov = __shfl_xor(bv, 1, 32);  oi = __shfl_xor(bi, 1, 32);
        if (ov > bv || (ov == bv && oi < bi)) { bv = ov; bi = oi; }
        last = bi;
        if (lane == 0) fpsi[b * 256 + it] = last;
    }
}

// ---------------------------------------------------------------------------
// K2: ball query; one wave per (b,p); ballot/popc stream compaction keeps
// the first 512 in-radius indices in ascending order (matches sorted select).
// ---------------------------------------------------------------------------
__global__ __launch_bounds__(256) void k2_ballq(const float* __restrict__ xyz,
                                                const int* __restrict__ fpsi,
                                                float* __restrict__ newxyz,
                                                int* __restrict__ idx)
{
    const int t    = blockIdx.x * 8 + (threadIdx.x >> 5);
    const int lane = threadIdx.x & 31;
    const int b    = t >> 8;
    const float* xb = xyz + b * 3072;
    const int j0 = fpsi[t];
    const float cx = xb[j0 * 3 + 0], cy = xb[j0 * 3 + 1], cz = xb[j0 * 3 + 2];
    if (lane < 3) newxyz[t * 3 + lane] = xb[j0 * 3 + lane];
    int* row = idx + (size_t)t * 512;
    int cnt = 0, first = 0;
    for (int c = 0; c < 32 && cnt < 512; ++c) {
        const int k = c * 32 + lane;
        float dx = xb[k * 3 + 0] - cx, dy = xb[k * 3 + 1] - cy, dz = xb[k * 3 + 2] - cz;
        float d2 = dx * dx + dy * dy + dz * dz;
        bool inr = d2 < RAD2;
        unsigned mask = (unsigned)__ballot(inr);
        if (cnt == 0 && mask) first = c * 32 + (__ffs(mask) - 1);
        int pos = cnt + __popc(mask & ((1u << lane) - 1u));
        if (inr && pos < 512) row[pos] = k;
        cnt += __popc(mask);
    }
    for (int q = cnt + lane; q < 512; q += 32) row[q] = first;
}

// ---------------------------------------------------------------------------
// K3: fused gather + 3-layer MLP (WMMA f16->f32) + max over samples.
// One workgroup (8 waves) per (b,p); 4 chunks of 128 sample-columns.
// ---------------------------------------------------------------------------
__global__ __launch_bounds__(256) void k3_mlp(
    const float* __restrict__ xyz, const half_t* __restrict__ featT,
    const half_t* __restrict__ w0p, const half_t* __restrict__ w1h, const half_t* __restrict__ w2h,
    const float* __restrict__ ab, const int* __restrict__ idx,
    const float* __restrict__ newxyz, float* __restrict__ feat)
{
    __shared__ half_t sX[128 * 296];   // X (stride 296); reused as H2 (stride 136)
    __shared__ half_t sH1[128 * 136];
    __shared__ int    sidx[512];
    __shared__ float  sab[768];
    __shared__ float  snew[3];

    const int tid   = threadIdx.x;
    const int lane  = tid & 31;
    const int wave  = tid >> 5;
    const int group = lane >> 4;
    const int l16   = lane & 15;
    const int wbase = wave * 16;
    const int wg    = blockIdx.x;
    const int b     = wg >> 8;

    for (int i = tid; i < 512; i += 256) sidx[i] = idx[(size_t)wg * 512 + i];
    for (int i = tid; i < 768; i += 256) sab[i] = ab[i];
    if (tid < 3) snew[tid] = newxyz[wg * 3 + tid];
    __syncthreads();

    float maxp[8];
#pragma unroll
    for (int r = 0; r < 8; ++r) maxp[r] = -1e30f;

    const float* xb = xyz + b * 3072;

    for (int c4 = 0; c4 < 4; ++c4) {
        // --- gather: 2 threads per column; rows 0..255 features, 256..258 xyz, 259..287 zero
        {
            const int col = tid >> 1;
            const int hh  = tid & 1;
            const int j   = sidx[c4 * 128 + col];
            const v8h* src = (const v8h*)(featT + (((size_t)b * 1024 + j) * 256 + hh * 128));
            v8h* dst = (v8h*)(sX + col * 296 + hh * 128);
#pragma unroll
            for (int i = 0; i < 16; ++i) dst[i] = src[i];
            if (hh == 0) {
                float px = xb[j * 3 + 0], py = xb[j * 3 + 1], pz = xb[j * 3 + 2];
                half_t hz = (half_t)0.0f;
                v8h z = {hz, hz, hz, hz, hz, hz, hz, hz};
                v8h g = z;
                g[0] = (half_t)((px - snew[0]) / 5.0f);
                g[1] = (half_t)((py - snew[1]) / 5.0f);
                g[2] = (half_t)((pz - snew[2]) / 5.0f);
                v8h* d2v = (v8h*)(sX + col * 296 + 256);
                d2v[0] = g; d2v[1] = z; d2v[2] = z; d2v[3] = z;
            }
        }
        __syncthreads();

        // --- layer0: X(296, K=288) -> H1(136)
        {
            v8f acc[8];
#pragma unroll
            for (int nt = 0; nt < 8; ++nt) acc[nt] = vzero8();
            gemm_tiles(w0p, 288, 9, sX, 296, wbase, lane, acc);
            epi_store(acc, &sab[0], &sab[128], sH1, 136, wbase, lane);
        }
        __syncthreads();

        // --- layer1: H1(136, K=128) -> H2(=sX, 136)
        {
            v8f acc[8];
#pragma unroll
            for (int nt = 0; nt < 8; ++nt) acc[nt] = vzero8();
            gemm_tiles(w1h, 128, 4, sH1, 136, wbase, lane, acc);
            epi_store(acc, &sab[256], &sab[384], sX, 136, wbase, lane);
        }
        __syncthreads();

        // --- layer2: H2(136, K=128) -> running max in registers
        {
            v8f acc[8];
#pragma unroll
            for (int nt = 0; nt < 8; ++nt) acc[nt] = vzero8();
            gemm_tiles(w2h, 128, 4, sX, 136, wbase, lane, acc);
            const int ch0 = wbase + group * 8;
#pragma unroll
            for (int nt = 0; nt < 8; ++nt) {
#pragma unroll
                for (int r = 0; r < 8; ++r) {
                    float y = acc[nt][r] * sab[512 + ch0 + r] + sab[640 + ch0 + r];
                    y = fmaxf(y, 0.0f);
                    maxp[r] = fmaxf(maxp[r], y);
                }
            }
        }
        __syncthreads();
    }

#pragma unroll
    for (int r = 0; r < 8; ++r) {
        maxp[r] = fmaxf(maxp[r], __shfl_xor(maxp[r], 1, 32));
        maxp[r] = fmaxf(maxp[r], __shfl_xor(maxp[r], 2, 32));
        maxp[r] = fmaxf(maxp[r], __shfl_xor(maxp[r], 4, 32));
        maxp[r] = fmaxf(maxp[r], __shfl_xor(maxp[r], 8, 32));
    }
    if (l16 == 0) {
        float* fo = feat + (size_t)wg * 128 + wbase + group * 8;
#pragma unroll
        for (int r = 0; r < 8; ++r) fo[r] = maxp[r];
    }
}

// ---------------------------------------------------------------------------
// K4: per-point head (2x 128->128 WMMA layers + 93x128 final) + output assembly.
// 8 workgroups x 128 point-columns.
// Output layout per point: [center(3) | size_scores(18) | srn(54) | sr(54) | sem(18)]
// ---------------------------------------------------------------------------
__global__ __launch_bounds__(256) void k4_head(
    const float* __restrict__ feat, const half_t* __restrict__ pw0h,
    const half_t* __restrict__ pw1h, const half_t* __restrict__ pw2p,
    const float* __restrict__ ab, const float* __restrict__ pb2p,
    const float* __restrict__ msd, const float* __restrict__ newxyz,
    float* __restrict__ out)
{
    __shared__ half_t sX[128 * 136];   // X; reused as H2
    __shared__ half_t sH1[128 * 136];
    __shared__ float  sab2[512];
    __shared__ float  spb2[96];
    __shared__ float  sms[54];

    const int tid   = threadIdx.x;
    const int lane  = tid & 31;
    const int wave  = tid >> 5;
    const int group = lane >> 4;
    const int l16   = lane & 15;
    const int wbase = wave * 16;
    const int colBase = blockIdx.x * 128;

    for (int i = tid; i < 512; i += 256) sab2[i] = ab[768 + i];
    if (tid < 96) spb2[tid] = pb2p[tid];
    if (tid < 54) sms[tid] = msd[tid];
    {
        const int col = tid >> 1;
        const int hh  = tid & 1;
        const float4* s4 = (const float4*)(feat + ((size_t)(colBase + col)) * 128 + hh * 64);
        v8h* d8 = (v8h*)(sX + col * 136 + hh * 64);
#pragma unroll
        for (int i = 0; i < 8; ++i) {
            float4 u = s4[2 * i], w = s4[2 * i + 1];
            v8h o = {(half_t)u.x, (half_t)u.y, (half_t)u.z, (half_t)u.w,
                     (half_t)w.x, (half_t)w.y, (half_t)w.z, (half_t)w.w};
            d8[i] = o;
        }
    }
    __syncthreads();

    { // pw0: X -> H1
        v8f acc[8];
#pragma unroll
        for (int nt = 0; nt < 8; ++nt) acc[nt] = vzero8();
        gemm_tiles(pw0h, 128, 4, sX, 136, wbase, lane, acc);
        epi_store(acc, &sab2[0], &sab2[128], sH1, 136, wbase, lane);
    }
    __syncthreads();
    { // pw1: H1 -> H2(=sX)
        v8f acc[8];
#pragma unroll
        for (int nt = 0; nt < 8; ++nt) acc[nt] = vzero8();
        gemm_tiles(pw1h, 128, 4, sH1, 136, wbase, lane, acc);
        epi_store(acc, &sab2[256], &sab2[384], sX, 136, wbase, lane);
    }
    __syncthreads();
    // final 96x128 (rows >=93 are zero-padded and skipped); waves 0..5 only
    if (wbase < 96) {
        v8f acc[8];
#pragma unroll
        for (int nt = 0; nt < 8; ++nt) acc[nt] = vzero8();
        gemm_tiles(pw2p, 128, 4, sX, 136, wbase, lane, acc);
        const int ch0 = wbase + group * 8;
#pragma unroll
        for (int nt = 0; nt < 8; ++nt) {
            const int gcol = colBase + nt * 16 + l16;
            float* ob = out + (size_t)gcol * 147;
#pragma unroll
            for (int r = 0; r < 8; ++r) {
                const int co = ch0 + r;
                if (co < 93) {
                    float y = acc[nt][r] + spb2[co];
                    if (co < 3)       ob[co] = y + newxyz[gcol * 3 + co];
                    else if (co < 21) ob[co] = y;
                    else if (co < 75) { ob[co] = y; ob[co + 54] = y * sms[co - 21]; }
                    else              ob[co + 54] = y;
                }
            }
        }
    }
}

// ---------------------------------------------------------------------------
extern "C" void kernel_launch(void* const* d_in, const int* in_sizes, int n_in,
                              void* d_out, int out_size, void* d_ws, size_t ws_size,
                              hipStream_t stream)
{
    (void)in_sizes; (void)n_in; (void)out_size; (void)ws_size;
    const float* xyz      = (const float*)d_in[0];
    const float* features = (const float*)d_in[1];
    const float* w0  = (const float*)d_in[2];
    const float* g0  = (const float*)d_in[3];
    const float* b0  = (const float*)d_in[4];
    const float* m0  = (const float*)d_in[5];
    const float* v0  = (const float*)d_in[6];
    const float* w1  = (const float*)d_in[7];
    const float* g1  = (const float*)d_in[8];
    const float* b1  = (const float*)d_in[9];
    const float* m1  = (const float*)d_in[10];
    const float* v1  = (const float*)d_in[11];
    const float* w2  = (const float*)d_in[12];
    const float* g2  = (const float*)d_in[13];
    const float* b2  = (const float*)d_in[14];
    const float* m2  = (const float*)d_in[15];
    const float* v2  = (const float*)d_in[16];
    const float* pw0 = (const float*)d_in[17];
    const float* pg0 = (const float*)d_in[18];
    const float* pb0 = (const float*)d_in[19];
    const float* pm0 = (const float*)d_in[20];
    const float* pv0 = (const float*)d_in[21];
    const float* pw1 = (const float*)d_in[22];
    const float* pg1 = (const float*)d_in[23];
    const float* pb1 = (const float*)d_in[24];
    const float* pm1 = (const float*)d_in[25];
    const float* pv1 = (const float*)d_in[26];
    const float* pw2 = (const float*)d_in[27];
    const float* pb2 = (const float*)d_in[28];
    const float* ms  = (const float*)d_in[29];

    char* ws = (char*)d_ws;
    half_t* featT = (half_t*)(ws + 0);          // 2,097,152 B
    half_t* w0p   = (half_t*)(ws + 2097152);    //    73,728
    half_t* w1h   = (half_t*)(ws + 2170880);    //    32,768
    half_t* w2h   = (half_t*)(ws + 2203648);    //    32,768
    half_t* pw0h  = (half_t*)(ws + 2236416);    //    32,768
    half_t* pw1h  = (half_t*)(ws + 2269184);    //    32,768
    half_t* pw2p  = (half_t*)(ws + 2301952);    //    24,576
    float*  ab    = (float*)(ws + 2326528);     //     5,120
    float*  pb2p  = (float*)(ws + 2331648);     //       512
    float*  msd   = (float*)(ws + 2332160);     //       256
    int*    fpsi  = (int*)(ws + 2332416);       //     4,096
    float*  nxyz  = (float*)(ws + 2336512);     //    12,288
    int*    idx   = (int*)(ws + 2348800);       // 2,097,152
    float*  feat  = (float*)(ws + 4445952);     //   524,288

    k0_featT<<<4096, 256, 0, stream>>>(features, featT);
    k0_misc<<<1, 256, 0, stream>>>(w0, w1, w2, pw0, pw1, pw2, pb2,
                                   g0, b0, m0, v0, g1, b1, m1, v1, g2, b2, m2, v2,
                                   pg0, pb0, pm0, pv0, pg1, pb1, pm1, pv1, ms,
                                   w0p, w1h, w2h, pw0h, pw1h, pw2p, ab, pb2p, msd);
    k1_fps<<<4, 32, 0, stream>>>(xyz, fpsi);
    k2_ballq<<<128, 256, 0, stream>>>(xyz, fpsi, nxyz, idx);
    k3_mlp<<<1024, 256, 0, stream>>>(xyz, featT, w0p, w1h, w2h, ab, idx, nxyz, feat);
    k4_head<<<8, 256, 0, stream>>>(feat, pw0h, pw1h, pw2p, ab, pb2p, msd, nxyz,
                                   (float*)d_out);
}
